// LocalFeatureAggregation_11003706212691
// MI455X (gfx1250) — compile-verified, matrix operand order checked by reading, not run on hardware
//
#include <hip/hip_runtime.h>
#include <hip/hip_bf16.h>
#include <math.h>

// ---------------------------------------------------------------------------
// Problem constants (from reference setup_inputs)
// ---------------------------------------------------------------------------
#define B_   4
#define N_   8192
#define C_   32
#define K_   16
#define H_   64
#define OUT_ 128
#define R_   (B_ * N_)        // 32768 points
#define RK_  (R_ * K_)        // 524288 (point, neighbor) rows

typedef __attribute__((ext_vector_type(16))) _Float16 v16h;
typedef __attribute__((ext_vector_type(8)))  float    v8f;

// ---------------------------------------------------------------------------
// WMMA fragment load: 16-bit A/B layout for v_wmma_f32_16x16x32_f16.
// For a lane, the 16 f16 elements are two contiguous 8-element (16B) runs:
//   elems 0..7  = row[ks .. ks+7]
//   elems 8..15 = row[ks+16 .. ks+23]
// where ks = kstep*32 + (lane>=16 ? 8 : 0).
// ---------------------------------------------------------------------------
__device__ __forceinline__ v16h load_frag(const _Float16* p) {
    union { uint4 q[2]; v16h v; } u;
    u.q[0] = *(const uint4*)(p);
    u.q[1] = *(const uint4*)(p + 16);
    return u.v;
}

// ---------------------------------------------------------------------------
// Generic GEMM + BN-stat kernel.
//   A : [M, KA] f16 row-major (KA = padded reduction dim, multiple of 32)
//   W : [NT*16, KA] f16 row-major (weight[o][k]) -> computes A * W^T
//   out : [M, NT*16] f32 (pre-BN)
//   gsum/gsq : per-channel sum / sum-of-squares accumulators (pre-zeroed)
// Block = 256 threads = 8 waves; each wave does one 16-row M tile x all N tiles.
// ---------------------------------------------------------------------------
template<int NT, int KA>
__global__ __launch_bounds__(256) void gemm_bn_kernel(
    const _Float16* __restrict__ A,
    const _Float16* __restrict__ W,
    int M,
    float* __restrict__ out,
    float* __restrict__ gsum,
    float* __restrict__ gsq)
{
    __shared__ __align__(16) _Float16 sW[NT * 16 * KA];
    __shared__ float lsum[NT * 16];
    __shared__ float lsq [NT * 16];

    const int tid = threadIdx.x;

    // Cooperative load of the full weight tile into LDS (16B chunks).
    {
        const uint4* src = (const uint4*)W;
        uint4* dst = (uint4*)sW;
        const int chunks = NT * 16 * KA / 8;
        for (int i = tid; i < chunks; i += 256) dst[i] = src[i];
    }
    if (tid < NT * 16) { lsum[tid] = 0.f; lsq[tid] = 0.f; }
    __syncthreads();

    const int wave  = tid >> 5;
    const int lane  = tid & 31;
    const int laneN = lane & 15;
    const int hi    = lane >> 4;        // 0: lanes 0-15, 1: lanes 16-31
    const int mBase = (blockIdx.x * 8 + wave) * 16;

    if (mBase < M) {
        v8f acc[NT] = {};
        const _Float16* Arow = A + (size_t)(mBase + laneN) * KA;

        #pragma unroll
        for (int kk = 0; kk < KA; kk += 32) {
            const int ks = kk + hi * 8;
            v16h a = load_frag(Arow + ks);
            #pragma unroll
            for (int nt = 0; nt < NT; ++nt) {
                v16h b = load_frag(sW + (size_t)(nt * 16 + laneN) * KA + ks);
                acc[nt] = __builtin_amdgcn_wmma_f32_16x16x32_f16(
                    false, a, false, b, (short)0, acc[nt], false, false);
            }
        }

        const int Nout = NT * 16;
        #pragma unroll
        for (int nt = 0; nt < NT; ++nt) {
            float s = 0.f, s2 = 0.f;
            #pragma unroll
            for (int r = 0; r < 8; ++r) {
                float v = acc[nt][r];
                const int m = mBase + r + hi * 8;   // C/D layout: VGPR r -> M=r (+8 for hi lanes)
                out[(size_t)m * Nout + nt * 16 + laneN] = v;
                s  += v;
                s2 += v * v;
            }
            atomicAdd(&lsum[nt * 16 + laneN], s);
            atomicAdd(&lsq [nt * 16 + laneN], s2);
        }
    }
    __syncthreads();
    if (tid < NT * 16) {
        atomicAdd(&gsum[tid], lsum[tid]);
        atomicAdd(&gsq [tid], lsq [tid]);
    }
}

// ---------------------------------------------------------------------------
// Brute-force KNN: each thread owns one query, keeps a sorted top-16 in
// registers, candidates staged through LDS in tiles of 256.
// ---------------------------------------------------------------------------
__global__ __launch_bounds__(256) void knn_kernel(const float* __restrict__ xyz,
                                                  int* __restrict__ idx_out)
{
    const int TS = 256;
    __shared__ float sx[TS], sy[TS], sz[TS];

    const int b = blockIdx.x >> 5;                        // N/256 = 32 blocks per batch
    const int n = ((blockIdx.x & 31) << 8) + threadIdx.x;
    const float* base = xyz + (size_t)b * N_ * 3;

    const float qx = base[n * 3 + 0];
    const float qy = base[n * 3 + 1];
    const float qz = base[n * 3 + 2];

    float bd[K_]; int bi[K_];
    #pragma unroll
    for (int i = 0; i < K_; ++i) { bd[i] = 3.4e38f; bi[i] = 0; }

    for (int t = 0; t < N_; t += TS) {
        const int c = t + threadIdx.x;
        sx[threadIdx.x] = base[c * 3 + 0];
        sy[threadIdx.x] = base[c * 3 + 1];
        sz[threadIdx.x] = base[c * 3 + 2];
        __syncthreads();
        for (int j = 0; j < TS; ++j) {
            const float dx = qx - sx[j];
            const float dy = qy - sy[j];
            const float dz = qz - sz[j];
            const float d = dx * dx + dy * dy + dz * dz;
            if (d < bd[K_ - 1]) {                        // bubble into sorted array
                bd[K_ - 1] = d; bi[K_ - 1] = t + j;
                #pragma unroll
                for (int i = K_ - 1; i > 0; --i) {
                    if (bd[i] < bd[i - 1]) {
                        float td = bd[i]; bd[i] = bd[i - 1]; bd[i - 1] = td;
                        int   ti = bi[i]; bi[i] = bi[i - 1]; bi[i - 1] = ti;
                    }
                }
            }
        }
        __syncthreads();
    }

    int* o = idx_out + ((size_t)b * N_ + n) * K_;
    #pragma unroll
    for (int i = 0; i < K_; ++i) o[i] = bi[i];
}

// ---------------------------------------------------------------------------
// Build LSE encoding [RK, 64] f16 (36 real cols: rel(3), dist(1), nbr_feat(32);
// padded with zeros to 64 so GEMM K is a multiple of 32).
// ---------------------------------------------------------------------------
__global__ void encode_kernel(const float* __restrict__ xyz,
                              const float* __restrict__ feats,
                              const int* __restrict__ idx,
                              _Float16* __restrict__ enc)
{
    const size_t g = (size_t)blockIdx.x * blockDim.x + threadIdx.x;
    if (g >= (size_t)RK_) return;
    const size_t bn = g / K_;
    const int n = (int)(bn % N_);
    const int b = (int)(bn / N_);
    const int p = idx[g];

    const float* q = xyz + ((size_t)b * N_ + n) * 3;
    const float* c = xyz + ((size_t)b * N_ + p) * 3;
    const float rx = c[0] - q[0], ry = c[1] - q[1], rz = c[2] - q[2];
    const float d = sqrtf(fmaxf(rx * rx + ry * ry + rz * rz, 1e-12f));

    __align__(16) _Float16 tmp[64];
    tmp[0] = (_Float16)rx; tmp[1] = (_Float16)ry;
    tmp[2] = (_Float16)rz; tmp[3] = (_Float16)d;
    const float* f = feats + ((size_t)b * N_ + p) * C_;
    #pragma unroll
    for (int i = 0; i < C_; ++i) tmp[4 + i] = (_Float16)f[i];
    #pragma unroll
    for (int i = 36; i < 64; ++i) tmp[i] = (_Float16)0.f;

    uint4* dst = (uint4*)(enc + g * 64);
    const uint4* src = (const uint4*)tmp;
    #pragma unroll
    for (int i = 0; i < 8; ++i) dst[i] = src[i];
}

// ---------------------------------------------------------------------------
// Small helpers
// ---------------------------------------------------------------------------
__global__ void zero_kernel(float* p, int n) {
    int g = blockIdx.x * blockDim.x + threadIdx.x;
    if (g < n) p[g] = 0.f;
}

__global__ void cast_w_kernel(const float* __restrict__ w, _Float16* __restrict__ wh,
                              int O, int Kin, int Kpad)
{
    int g = blockIdx.x * blockDim.x + threadIdx.x;
    if (g >= O * Kpad) return;
    const int k = g % Kpad, o = g / Kpad;
    wh[g] = (k < Kin) ? (_Float16)w[o * Kin + k] : (_Float16)0.f;
}

__global__ void bn_finalize_kernel(float* sum, float* sq, int C, float invCnt) {
    int t = blockIdx.x * blockDim.x + threadIdx.x;
    if (t >= C) return;
    const float m = sum[t] * invCnt;
    const float var = sq[t] * invCnt - m * m;
    sum[t] = m;                        // reuse buffers: sum -> mean
    sq[t]  = rsqrtf(var + 1e-5f);      //                sq  -> rstd
}

// Apply BN (+optional ReLU), write f16 (and optionally f32) with stride/offset.
__global__ void bn_apply_kernel(const float* __restrict__ in,
                                const float* __restrict__ mean,
                                const float* __restrict__ rstd,
                                int C, size_t total,
                                _Float16* out_h, int out_stride, int col_off,
                                float* out_f, int relu)
{
    const size_t g = (size_t)blockIdx.x * blockDim.x + threadIdx.x;
    if (g >= total) return;
    const int c = (int)(g % (size_t)C);
    const size_t m = g / (size_t)C;
    float v = (in[g] - mean[c]) * rstd[c];
    if (relu) v = fmaxf(v, 0.f);
    const size_t o = m * (size_t)out_stride + col_off + c;
    if (out_h) out_h[o] = (_Float16)v;
    if (out_f) out_f[o] = v;
}

// Attention logits + softmax over k=16 (lane groups of 16 within a wave32).
__global__ __launch_bounds__(256) void attn_scores_kernel(
    const _Float16* __restrict__ h, const float* __restrict__ w2,
    const float* __restrict__ b2, float* __restrict__ scores, int R)
{
    const int g = blockIdx.x * blockDim.x + threadIdx.x;
    const int r = g >> 4;
    const int k = g & 15;
    if (r >= R) return;
    const _Float16* hr = h + ((size_t)r * 16 + k) * 64;
    float acc = 0.f;
    #pragma unroll
    for (int c = 0; c < 64; ++c) acc += (float)hr[c] * w2[c];
    acc += b2[0];
    float m = acc;
    #pragma unroll
    for (int s = 8; s >= 1; s >>= 1) m = fmaxf(m, __shfl_xor(m, s, 16));
    const float e = __expf(acc - m);
    float sum = e;
    #pragma unroll
    for (int s = 8; s >= 1; s >>= 1) sum += __shfl_xor(sum, s, 16);
    scores[g] = e / sum;
}

// feat[r, c] = sum_k x[r, k, c] * score[r, k]  -> f16 output
__global__ void feat_kernel(const _Float16* __restrict__ x,
                            const float* __restrict__ scores,
                            _Float16* __restrict__ feat, int R)
{
    const int g = blockIdx.x * blockDim.x + threadIdx.x;
    const int r = g >> 6;
    const int c = g & 63;
    if (r >= R) return;
    float acc = 0.f;
    #pragma unroll
    for (int k = 0; k < 16; ++k)
        acc += (float)x[((size_t)r * 16 + k) * 64 + c] * scores[r * 16 + k];
    feat[(size_t)r * 64 + c] = (_Float16)acc;
}

// out = relu(bn(in) + residual)   (C = 128)
__global__ void final_kernel(const float* __restrict__ in,
                             const float* __restrict__ mean,
                             const float* __restrict__ rstd,
                             const float* __restrict__ res,
                             float* __restrict__ out, size_t total)
{
    const size_t g = (size_t)blockIdx.x * blockDim.x + threadIdx.x;
    if (g >= total) return;
    const int c = (int)(g & 127);
    const float v = (in[g] - mean[c]) * rstd[c] + res[g];
    out[g] = fmaxf(v, 0.f);
}

// ---------------------------------------------------------------------------
// Host orchestration
// ---------------------------------------------------------------------------
extern "C" void kernel_launch(void* const* d_in, const int* in_sizes, int n_in,
                              void* d_out, int out_size, void* d_ws, size_t ws_size,
                              hipStream_t stream) {
    (void)in_sizes; (void)n_in; (void)out_size; (void)ws_size;

    const float* xyz   = (const float*)d_in[0];
    const float* feats = (const float*)d_in[1];
    const float* wl_f  [2] = { (const float*)d_in[2], (const float*)d_in[3]  };
    const float* sw1_f [2] = { (const float*)d_in[4], (const float*)d_in[8]  };
    const float* sw2_f [2] = { (const float*)d_in[5], (const float*)d_in[9]  };
    const float* sb2_f [2] = { (const float*)d_in[6], (const float*)d_in[10] };
    const float* mw_f  [2] = { (const float*)d_in[7], (const float*)d_in[11] };
    const float* d1_f = (const float*)d_in[12];
    const float* d2_f = (const float*)d_in[13];
    float* out = (float*)d_out;

    // ---- workspace carve-up (256B aligned) ----
    char* base = (char*)d_ws;
    size_t off = 0;
    auto alloc = [&](size_t bytes) -> void* {
        off = (off + 255) & ~(size_t)255;
        void* p = base + off;
        off += bytes;
        return p;
    };
    int*      idx    = (int*)     alloc((size_t)RK_ * 4);
    _Float16* enc    = (_Float16*)alloc((size_t)RK_ * 64 * 2);
    _Float16* x_h    = (_Float16*)alloc((size_t)RK_ * 64 * 2);
    _Float16* h_h    = (_Float16*)alloc((size_t)RK_ * 64 * 2);
    float*    tmpA   = (float*)   alloc((size_t)RK_ * 64 * 4);   // shared pre-BN staging
    float*    scores = (float*)   alloc((size_t)RK_ * 4);
    _Float16* feat_h = (_Float16*)alloc((size_t)R_ * 64 * 2);
    _Float16* agg_h  = (_Float16*)alloc((size_t)R_ * 128 * 2);
    float*    agg_f  = (float*)   alloc((size_t)R_ * 128 * 4);
    _Float16* h1_h   = (_Float16*)alloc((size_t)R_ * 128 * 2);
    _Float16* wlh[2] = { (_Float16*)alloc(64 * 64 * 2),  (_Float16*)alloc(64 * 64 * 2)  };
    _Float16* s1h[2] = { (_Float16*)alloc(64 * 64 * 2),  (_Float16*)alloc(64 * 64 * 2)  };
    _Float16* mwh[2] = { (_Float16*)alloc(64 * 64 * 2),  (_Float16*)alloc(64 * 64 * 2)  };
    _Float16* d1h = (_Float16*)alloc(128 * 128 * 2);
    _Float16* d2h = (_Float16*)alloc(128 * 128 * 2);
    float*    stats = (float*)alloc(256 * 4);                    // sum[128] ++ sumsq[128]
    float* ssum = stats;
    float* ssq  = stats + 128;

    const dim3 blk(256);

    // ---- weight cast / pad ----
    for (int br = 0; br < 2; ++br) {
        cast_w_kernel<<<dim3((64 * 64 + 255) / 256), blk, 0, stream>>>(wl_f[br], wlh[br], 64, 36, 64);
        cast_w_kernel<<<dim3((64 * 64 + 255) / 256), blk, 0, stream>>>(sw1_f[br], s1h[br], 64, 64, 64);
        cast_w_kernel<<<dim3((64 * 64 + 255) / 256), blk, 0, stream>>>(mw_f[br],  mwh[br], 64, 64, 64);
    }
    cast_w_kernel<<<dim3((128 * 128 + 255) / 256), blk, 0, stream>>>(d1_f, d1h, 128, 128, 128);
    cast_w_kernel<<<dim3((128 * 128 + 255) / 256), blk, 0, stream>>>(d2_f, d2h, 128, 128, 128);

    // ---- KNN + neighbor encoding (shared by both branches) ----
    knn_kernel<<<dim3(R_ / 256), blk, 0, stream>>>(xyz, idx);
    encode_kernel<<<dim3((RK_ + 255) / 256), blk, 0, stream>>>(xyz, feats, idx, enc);

    const size_t RK64 = (size_t)RK_ * 64;
    const size_t R64  = (size_t)R_ * 64;
    const size_t R128 = (size_t)R_ * 128;

    // ---- two attention branches ----
    for (int br = 0; br < 2; ++br) {
        // LSE: enc[RK,64(36)] x w_lse^T -> [RK,64], BN(0,1,2) + ReLU
        zero_kernel<<<dim3(1), blk, 0, stream>>>(stats, 256);
        gemm_bn_kernel<4, 64><<<dim3((RK_ + 127) / 128), blk, 0, stream>>>(enc, wlh[br], RK_, tmpA, ssum, ssq);
        bn_finalize_kernel<<<dim3(1), dim3(128), 0, stream>>>(ssum, ssq, 64, 1.f / (float)RK_);
        bn_apply_kernel<<<dim3((unsigned)((RK64 + 255) / 256)), blk, 0, stream>>>(
            tmpA, ssum, ssq, 64, RK64, x_h, 64, 0, nullptr, 1);

        // att_pool hidden: x[RK,64] x sw1^T -> BN + ReLU
        zero_kernel<<<dim3(1), blk, 0, stream>>>(stats, 256);
        gemm_bn_kernel<4, 64><<<dim3((RK_ + 127) / 128), blk, 0, stream>>>(x_h, s1h[br], RK_, tmpA, ssum, ssq);
        bn_finalize_kernel<<<dim3(1), dim3(128), 0, stream>>>(ssum, ssq, 64, 1.f / (float)RK_);
        bn_apply_kernel<<<dim3((unsigned)((RK64 + 255) / 256)), blk, 0, stream>>>(
            tmpA, ssum, ssq, 64, RK64, h_h, 64, 0, nullptr, 1);

        // softmax over k, weighted pooling
        attn_scores_kernel<<<dim3(RK_ / 256), blk, 0, stream>>>(h_h, sw2_f[br], sb2_f[br], scores, R_);
        feat_kernel<<<dim3((unsigned)((R64 + 255) / 256)), blk, 0, stream>>>(x_h, scores, feat_h, R_);

        // mlp: feat[R,64] x mw^T -> BN + ReLU -> agg[:, 64*br : 64*br+64]
        zero_kernel<<<dim3(1), blk, 0, stream>>>(stats, 256);
        gemm_bn_kernel<4, 64><<<dim3((R_ + 127) / 128), blk, 0, stream>>>(feat_h, mwh[br], R_, tmpA, ssum, ssq);
        bn_finalize_kernel<<<dim3(1), dim3(128), 0, stream>>>(ssum, ssq, 64, 1.f / (float)R_);
        bn_apply_kernel<<<dim3((unsigned)((R64 + 255) / 256)), blk, 0, stream>>>(
            tmpA, ssum, ssq, 64, R64, agg_h, 128, 64 * br, agg_f, 1);
    }

    // ---- dilated residual block on agg[R,128] ----
    zero_kernel<<<dim3(1), blk, 0, stream>>>(stats, 256);
    gemm_bn_kernel<8, 128><<<dim3((R_ + 127) / 128), blk, 0, stream>>>(agg_h, d1h, R_, tmpA, ssum, ssq);
    bn_finalize_kernel<<<dim3(1), dim3(128), 0, stream>>>(ssum, ssq, 128, 1.f / (float)R_);
    bn_apply_kernel<<<dim3((unsigned)((R128 + 255) / 256)), blk, 0, stream>>>(
        tmpA, ssum, ssq, 128, R128, h1_h, 128, 0, nullptr, 1);

    zero_kernel<<<dim3(1), blk, 0, stream>>>(stats, 256);
    gemm_bn_kernel<8, 128><<<dim3((R_ + 127) / 128), blk, 0, stream>>>(h1_h, d2h, R_, tmpA, ssum, ssq);
    bn_finalize_kernel<<<dim3(1), dim3(128), 0, stream>>>(ssum, ssq, 128, 1.f / (float)R_);
    final_kernel<<<dim3((unsigned)((R128 + 255) / 256)), blk, 0, stream>>>(
        tmpA, ssum, ssq, agg_f, out, R128);
}
